// CombinedGNNModel_51161650430391
// MI455X (gfx1250) — matrix operand order repaired and checked by reading, not compiled, not run on hardware
//
#include <hip/hip_runtime.h>
#include <hip/hip_bf16.h>
#include <hip/hip_fp16.h>

typedef __attribute__((ext_vector_type(16))) _Float16 v16h;
typedef __attribute__((ext_vector_type(8)))  float    v8f;

#define EPS_BN 1e-5f
#define CDIV(a, b) (((a) + (b) - 1) / (b))
#define USE_ASYNC_LDS 1

// ---------------------------------------------------------------------------
// Async weight staging: global f16 weights -> LDS, via GLOBAL_LOAD_ASYNC_TO_LDS
// (ASYNCcnt-tracked DMA per CDNA5 ISA). Each of the 256
// threads issues one b128 (8 halves) per pass. Wait on own ASYNCcnt, then
// workgroup barrier covers the other waves' transfers.
// ---------------------------------------------------------------------------
static __device__ __forceinline__ void async_copy_w(const _Float16* __restrict__ g,
                                                    _Float16* l, int numel) {
#if USE_ASYNC_LDS
  for (int i = threadIdx.x * 8; i < numel; i += 2048) {
    unsigned int lds_off = (unsigned int)(uintptr_t)(l + i);
    unsigned long long gaddr = (unsigned long long)(uintptr_t)(g + i);
    asm volatile("global_load_async_to_lds_b128 %0, %1, off"
                 :: "v"(lds_off), "v"(gaddr) : "memory");
  }
#else
  for (int i = threadIdx.x * 8; i < numel; i += 2048) {
    *(ulonglong2*)(l + i) = *(const ulonglong2*)(g + i);
  }
#endif
}

static __device__ __forceinline__ void wait_staging() {
#if USE_ASYNC_LDS
  asm volatile("s_wait_asynccnt 0x0" ::: "memory");
#endif
  __syncthreads();
}

// ---------------------------------------------------------------------------
// WMMA fragment loaders (wave32, v_wmma_f32_16x16x32_f16; A/B use this code's contiguous k order)
// A 16x32 f16: lane holds row m=lane&15; 16 contiguous K at kb + (lane>=16?16:0)
//              (same k order as B)
// B 32x16 f16: lane holds col n=lane&15; 16 contiguous K at kb + (lane>=16?16:0)
//              -> from f16 weights stored transposed [Hout][K]: ONE v16h read
// C/D 16x16 f32: element v of lane = (M = v + 8*(lane>>4), N = lane&15)
// ---------------------------------------------------------------------------
static __device__ __forceinline__ v16h load_a_frag(const float* __restrict__ A, int lda,
                                                   int row, int kbase, int lane, float scale) {
  const float* p = A + (size_t)row * lda + kbase + ((lane >> 4) << 4);
  const float4 x0 = *(const float4*)(p + 0);
  const float4 x1 = *(const float4*)(p + 4);
  const float4 x2 = *(const float4*)(p + 8);
  const float4 x3 = *(const float4*)(p + 12);
  v16h a;
  a[0]  = (_Float16)(x0.x * scale); a[1]  = (_Float16)(x0.y * scale);
  a[2]  = (_Float16)(x0.z * scale); a[3]  = (_Float16)(x0.w * scale);
  a[4]  = (_Float16)(x1.x * scale); a[5]  = (_Float16)(x1.y * scale);
  a[6]  = (_Float16)(x1.z * scale); a[7]  = (_Float16)(x1.w * scale);
  a[8]  = (_Float16)(x2.x * scale); a[9]  = (_Float16)(x2.y * scale);
  a[10] = (_Float16)(x2.z * scale); a[11] = (_Float16)(x2.w * scale);
  a[12] = (_Float16)(x3.x * scale); a[13] = (_Float16)(x3.y * scale);
  a[14] = (_Float16)(x3.z * scale); a[15] = (_Float16)(x3.w * scale);
  return a;
}

static __device__ __forceinline__ v16h load_b_frag(const _Float16* __restrict__ Wt, int K,
                                                   int col, int kbase, int lane) {
  return *(const v16h*)(Wt + (size_t)col * K + kbase + ((lane >> 4) << 4));
}

// ---------------------------------------------------------------------------
// Small utility kernels
// ---------------------------------------------------------------------------
__global__ void k_convert_wt(const float* __restrict__ W, _Float16* __restrict__ Wt,
                             int K, int H) {
  int i = blockIdx.x * blockDim.x + threadIdx.x;
  if (i >= K * H) return;
  int k = i / H, c = i % H;
  Wt[(size_t)c * K + k] = (_Float16)W[i];
}

__global__ void k_zero(float* __restrict__ p, int n) {
  int i = blockIdx.x * blockDim.x + threadIdx.x;
  if (i < n) p[i] = 0.0f;
}

__global__ void k_edge_deg(const int* __restrict__ dst, float* __restrict__ cntE, int E) {
  int e = blockIdx.x * blockDim.x + threadIdx.x;
  if (e < E) atomicAdd(&cntE[dst[e]], 1.0f);
}

__global__ void k_finalize_deg(const float* __restrict__ cntE, float* __restrict__ dinv,
                               float* __restrict__ cinv, int N) {
  int i = blockIdx.x * blockDim.x + threadIdx.x;
  if (i >= N) return;
  float c = cntE[i];
  dinv[i] = rsqrtf(c + 1.0f);          // GCN: deg = in-degree + self loop
  cinv[i] = 1.0f / fmaxf(c, 1.0f);     // SAGE: mean divisor
}

// Scatter: acc[dst] += feat[src]  (64 features, float4 lanes, f32 atomics)
__global__ void k_edge_scatter(const float* __restrict__ feat, float* __restrict__ acc,
                               const int* __restrict__ src, const int* __restrict__ dst, int E) {
  int id = blockIdx.x * blockDim.x + threadIdx.x;
  int e = id >> 4;
  if (e >= E) return;
  int f = (id & 15) << 2;
  int s = src[e], d = dst[e];
  const float4 v = *(const float4*)(feat + (size_t)s * 64 + f);
  float* dp = acc + (size_t)d * 64 + f;
  atomicAdd(dp + 0, v.x); atomicAdd(dp + 1, v.y);
  atomicAdd(dp + 2, v.z); atomicAdd(dp + 3, v.w);
}

// ---------------------------------------------------------------------------
// WMMA GEMM kernels. One wave per 16x16 output tile; 8 waves / 256-thr block.
// Weights async-staged to LDS once per block; B-fragments come from LDS.
// ---------------------------------------------------------------------------
__global__ __launch_bounds__(256)
void k_gemm_in(const float* __restrict__ X, const _Float16* __restrict__ Wt,
               const float* __restrict__ bias, float* __restrict__ Y, int N, int K) {
  __shared__ __align__(32) _Float16 lw[128 * 64];  // K=128 max
  async_copy_w(Wt, lw, K * 64);
  wait_staging();

  int lane = threadIdx.x & 31;
  int wave = (blockIdx.x * blockDim.x + threadIdx.x) >> 5;
  int nrt  = (N + 15) >> 4;
  int row_tile = wave >> 2, col_tile = wave & 3;
  if (row_tile >= nrt) return;
  int row0 = row_tile << 4;
  int m = lane & 15;
  int arow = min(row0 + m, N - 1);
  v8f c = {};
  for (int kb = 0; kb < K; kb += 32) {
    v16h a = load_a_frag(X, K, arow, kb, lane, 1.0f);
    v16h b = load_b_frag(lw, K, (col_tile << 4) + m, kb, lane);
    c = __builtin_amdgcn_wmma_f32_16x16x32_f16(false, a, false, b, (short)0, c, false, false);
  }
  int colg = (col_tile << 4) + m;
  float bv = bias[colg];
#pragma unroll
  for (int v = 0; v < 8; ++v) {
    int rowg = row0 + v + ((lane >> 4) << 3);
    if (rowg < N) Y[(size_t)rowg * 64 + colg] = fmaxf(c[v] + bv, 0.0f);
  }
}

// GCN transform: u = (X @ W) * dinv[row]; write u AND acc=u (self-loop seed)
__global__ __launch_bounds__(256)
void k_gemm_gcn(const float* __restrict__ X, const _Float16* __restrict__ Wt,
                const float* __restrict__ dinv, float* __restrict__ U,
                float* __restrict__ Acc, int N) {
  const int K = 64;
  __shared__ __align__(32) _Float16 lw[64 * 64];
  async_copy_w(Wt, lw, K * 64);
  wait_staging();

  int lane = threadIdx.x & 31;
  int wave = (blockIdx.x * blockDim.x + threadIdx.x) >> 5;
  int nrt  = (N + 15) >> 4;
  int row_tile = wave >> 2, col_tile = wave & 3;
  if (row_tile >= nrt) return;
  int row0 = row_tile << 4;
  int m = lane & 15;
  int arow = min(row0 + m, N - 1);
  v8f c = {};
#pragma unroll
  for (int kb = 0; kb < K; kb += 32) {
    v16h a = load_a_frag(X, K, arow, kb, lane, 1.0f);
    v16h b = load_b_frag(lw, K, (col_tile << 4) + m, kb, lane);
    c = __builtin_amdgcn_wmma_f32_16x16x32_f16(false, a, false, b, (short)0, c, false, false);
  }
  int colg = (col_tile << 4) + m;
#pragma unroll
  for (int v = 0; v < 8; ++v) {
    int rowg = row0 + v + ((lane >> 4) << 3);
    if (rowg < N) {
      float val = c[v] * dinv[rowg];
      size_t idx = (size_t)rowg * 64 + colg;
      U[idx] = val;
      Acc[idx] = val;
    }
  }
}

// GCN epilogue: out = relu(bn(dinv[n]*acc + b))
__global__ void k_gcn_epi(const float* __restrict__ acc, const float* __restrict__ dinv,
                          const float* __restrict__ bias, const float* __restrict__ bn,
                          float* __restrict__ out, int N) {
  int id = blockIdx.x * blockDim.x + threadIdx.x;
  int n = id >> 6;
  if (n >= N) return;
  int f = id & 63;
  float val = dinv[n] * acc[id] + bias[f];
  val = (val - bn[128 + f]) * rsqrtf(bn[192 + f] + EPS_BN) * bn[f] + bn[64 + f];
  out[id] = fmaxf(val, 0.0f);
}

// SAGE fused: out = relu(bn( (ssum*cinv[row]) @ Wl + Hin @ Wr + b ))
__global__ __launch_bounds__(256)
void k_gemm_sage(const float* __restrict__ S, const float* __restrict__ Hin,
                 const _Float16* __restrict__ Wlt, const _Float16* __restrict__ Wrt,
                 const float* __restrict__ cinv, const float* __restrict__ bias,
                 const float* __restrict__ bn, float* __restrict__ out, int N) {
  const int K = 64;
  __shared__ __align__(32) _Float16 lwl[64 * 64];
  __shared__ __align__(32) _Float16 lwr[64 * 64];
  async_copy_w(Wlt, lwl, K * 64);
  async_copy_w(Wrt, lwr, K * 64);
  wait_staging();

  int lane = threadIdx.x & 31;
  int wave = (blockIdx.x * blockDim.x + threadIdx.x) >> 5;
  int nrt  = (N + 15) >> 4;
  int row_tile = wave >> 2, col_tile = wave & 3;
  if (row_tile >= nrt) return;
  int row0 = row_tile << 4;
  int m = lane & 15;
  int arow = min(row0 + m, N - 1);
  float cs = cinv[arow];
  v8f c = {};
#pragma unroll
  for (int kb = 0; kb < K; kb += 32) {
    v16h a1 = load_a_frag(S, K, arow, kb, lane, cs);
    v16h b1 = load_b_frag(lwl, K, (col_tile << 4) + m, kb, lane);
    c = __builtin_amdgcn_wmma_f32_16x16x32_f16(false, a1, false, b1, (short)0, c, false, false);
    v16h a2 = load_a_frag(Hin, K, arow, kb, lane, 1.0f);
    v16h b2 = load_b_frag(lwr, K, (col_tile << 4) + m, kb, lane);
    c = __builtin_amdgcn_wmma_f32_16x16x32_f16(false, a2, false, b2, (short)0, c, false, false);
  }
  int colg = (col_tile << 4) + m;
  float bv = bias[colg];
  float g  = bn[colg], be = bn[64 + colg], rm = bn[128 + colg];
  float rs = rsqrtf(bn[192 + colg] + EPS_BN);
#pragma unroll
  for (int v = 0; v < 8; ++v) {
    int rowg = row0 + v + ((lane >> 4) << 3);
    if (rowg < N) {
      float val = (c[v] + bv - rm) * rs * g + be;
      out[(size_t)rowg * 64 + colg] = fmaxf(val, 0.0f);
    }
  }
}

// ---------------------------------------------------------------------------
// Pooling + head
// ---------------------------------------------------------------------------
__global__ void k_pool_scatter(const float* __restrict__ hg, const float* __restrict__ hs,
                               const int* __restrict__ batch, float* __restrict__ pooled,
                               float* __restrict__ gcnt, int N) {
  int id = blockIdx.x * blockDim.x + threadIdx.x;
  int n = id >> 5;
  if (n >= N) return;
  int q = id & 31;
  int b = batch[n];
  float* pb = pooled + (size_t)b * 128;
  if (q < 16) {
    int f = q << 2;
    float4 v = *(const float4*)(hg + (size_t)n * 64 + f);
    atomicAdd(pb + f + 0, v.x); atomicAdd(pb + f + 1, v.y);
    atomicAdd(pb + f + 2, v.z); atomicAdd(pb + f + 3, v.w);
    if (q == 0) atomicAdd(&gcnt[b], 1.0f);
  } else {
    int f = (q - 16) << 2;
    float4 v = *(const float4*)(hs + (size_t)n * 64 + f);
    atomicAdd(pb + 64 + f + 0, v.x); atomicAdd(pb + 64 + f + 1, v.y);
    atomicAdd(pb + 64 + f + 2, v.z); atomicAdd(pb + 64 + f + 3, v.w);
  }
}

__global__ void k_pool_norm(float* __restrict__ pooled, const float* __restrict__ gcnt, int G) {
  int i = blockIdx.x * blockDim.x + threadIdx.x;
  if (i >= G * 128) return;
  pooled[i] /= fmaxf(gcnt[i >> 7], 1.0f);
}

__global__ void k_head(const float* __restrict__ pooled,
                       const float* __restrict__ fc1W, const float* __restrict__ fc1b,
                       const float* __restrict__ bn1,
                       const float* __restrict__ fc2W, const float* __restrict__ fc2b,
                       const float* __restrict__ bn2,
                       const float* __restrict__ fc3W, const float* __restrict__ fc3b,
                       float* __restrict__ out) {
  __shared__ float z1[64];
  __shared__ float z2[32];
  int g = blockIdx.x, f = threadIdx.x;  // 64 threads per block
  const float* p = pooled + (size_t)g * 128;
  float a1 = fc1b[f];
  for (int k = 0; k < 128; ++k) a1 += p[k] * fc1W[k * 64 + f];
  a1 = (a1 - bn1[128 + f]) * rsqrtf(bn1[192 + f] + EPS_BN) * bn1[f] + bn1[64 + f];
  z1[f] = fmaxf(a1, 0.0f);
  __syncthreads();
  if (f < 32) {
    float a2 = fc2b[f];
    for (int k = 0; k < 64; ++k) a2 += z1[k] * fc2W[k * 32 + f];
    a2 = (a2 - bn2[64 + f]) * rsqrtf(bn2[96 + f] + EPS_BN) * bn2[f] + bn2[32 + f];
    z2[f] = fmaxf(a2, 0.0f);
  }
  __syncthreads();
  if (f == 0) {
    float a3 = fc3b[0];
    for (int k = 0; k < 32; ++k) a3 += z2[k] * fc3W[k];
    out[g] = a3;
  }
}

// ---------------------------------------------------------------------------
// Host launch
// ---------------------------------------------------------------------------
extern "C" void kernel_launch(void* const* d_in, const int* in_sizes, int n_in,
                              void* d_out, int out_size, void* d_ws, size_t ws_size,
                              hipStream_t stream) {
  const float* x       = (const float*)d_in[0];
  const int*   edge    = (const int*)d_in[1];
  const int*   batch   = (const int*)d_in[2];
  const float* W_in    = (const float*)d_in[3];
  const float* b_in    = (const float*)d_in[4];
  const float* gcn_W   = (const float*)d_in[5];
  const float* gcn_b   = (const float*)d_in[6];
  const float* gcn_bn  = (const float*)d_in[7];
  const float* sage_Wl = (const float*)d_in[8];
  const float* sage_Wr = (const float*)d_in[9];
  const float* sage_b  = (const float*)d_in[10];
  const float* sage_bn = (const float*)d_in[11];
  const float* fc1_W   = (const float*)d_in[12];
  const float* fc1_b   = (const float*)d_in[13];
  const float* bn1     = (const float*)d_in[14];
  const float* fc2_W   = (const float*)d_in[15];
  const float* fc2_b   = (const float*)d_in[16];
  const float* bn2     = (const float*)d_in[17];
  const float* fc3_W   = (const float*)d_in[18];
  const float* fc3_b   = (const float*)d_in[19];

  const int D_IN = 128, H = 64, L = 2;
  const int N = in_sizes[0] / D_IN;
  const int E = in_sizes[1] / 2;
  const int G = out_size;
  const int* src = edge;
  const int* dst = edge + E;

  // ---- workspace carve (GCN buffers aliased for SAGE phase; phases are serial)
  char* wp = (char*)d_ws;
  auto take = [&](size_t bytes) -> char* {
    char* p = wp;
    wp += (bytes + 255) & ~(size_t)255;
    return p;
  };
  const size_t feat_b = (size_t)N * 64 * sizeof(float);
  float* h    = (float*)take(feat_b);
  float* bufA = (float*)take(feat_b);   // hg layer0 out  -> SAGE hs layer0 out
  float* hgB  = (float*)take(feat_b);   // hg final
  float* bufU = (float*)take(feat_b);   // GCN u          -> SAGE hs final
  float* bufC = (float*)take(feat_b);   // GCN acc        -> SAGE ssum
  float* cntE = (float*)take((size_t)N * sizeof(float));
  float* dinv = (float*)take((size_t)N * sizeof(float));
  float* cinv = (float*)take((size_t)N * sizeof(float));
  float* pooled = (float*)take((size_t)G * 128 * sizeof(float));
  float* gcnt   = (float*)take((size_t)G * sizeof(float));
  _Float16* WtIn  = (_Float16*)take((size_t)D_IN * H * sizeof(_Float16));
  _Float16* WtGcn = (_Float16*)take((size_t)L * H * H * sizeof(_Float16));
  _Float16* WtSl  = (_Float16*)take((size_t)L * H * H * sizeof(_Float16));
  _Float16* WtSr  = (_Float16*)take((size_t)L * H * H * sizeof(_Float16));

  const int TPB = 256;

  // ---- 1. weight conversion (f32 [K,H] -> f16 [H,K])
  k_convert_wt<<<CDIV(D_IN * H, TPB), TPB, 0, stream>>>(W_in, WtIn, D_IN, H);
  for (int l = 0; l < L; ++l) {
    k_convert_wt<<<CDIV(H * H, TPB), TPB, 0, stream>>>(gcn_W + l * H * H, WtGcn + l * H * H, H, H);
    k_convert_wt<<<CDIV(H * H, TPB), TPB, 0, stream>>>(sage_Wl + l * H * H, WtSl + l * H * H, H, H);
    k_convert_wt<<<CDIV(H * H, TPB), TPB, 0, stream>>>(sage_Wr + l * H * H, WtSr + l * H * H, H, H);
  }

  // ---- 2. degrees
  k_zero<<<CDIV(N, TPB), TPB, 0, stream>>>(cntE, N);
  k_edge_deg<<<CDIV(E, TPB), TPB, 0, stream>>>(dst, cntE, E);
  k_finalize_deg<<<CDIV(N, TPB), TPB, 0, stream>>>(cntE, dinv, cinv, N);

  // ---- 3. input transform: h = relu(x @ W_in + b_in)
  const int nrt = CDIV(N, 16);
  const int gemm_blocks = CDIV(nrt * 4, 8);  // 8 waves per block, 4 col tiles
  k_gemm_in<<<gemm_blocks, TPB, 0, stream>>>(x, WtIn, b_in, h, N, D_IN);

  // ---- 4. GCN branch
  {
    const float* in = h;
    float* outs[2] = {bufA, hgB};
    for (int l = 0; l < L; ++l) {
      k_gemm_gcn<<<gemm_blocks, TPB, 0, stream>>>(in, WtGcn + l * H * H, dinv, bufU, bufC, N);
      k_edge_scatter<<<CDIV(E * 16, TPB), TPB, 0, stream>>>(bufU, bufC, src, dst, E);
      k_gcn_epi<<<CDIV(N * 64, TPB), TPB, 0, stream>>>(bufC, dinv, gcn_b + l * H,
                                                       gcn_bn + l * 4 * H, outs[l], N);
      in = outs[l];
    }
  }

  // ---- 5. SAGE branch (reuses bufA as hs0 out, bufU as hs final, bufC as ssum)
  float* hsB = bufU;
  {
    const float* in = h;
    float* outs[2] = {bufA, hsB};
    for (int l = 0; l < L; ++l) {
      k_zero<<<CDIV(N * 64, TPB), TPB, 0, stream>>>(bufC, N * 64);
      k_edge_scatter<<<CDIV(E * 16, TPB), TPB, 0, stream>>>(in, bufC, src, dst, E);
      k_gemm_sage<<<gemm_blocks, TPB, 0, stream>>>(bufC, in, WtSl + l * H * H, WtSr + l * H * H,
                                                   cinv, sage_b + l * H, sage_bn + l * 4 * H,
                                                   outs[l], N);
      in = outs[l];
    }
  }

  // ---- 6. pooling
  k_zero<<<CDIV(G * 128, TPB), TPB, 0, stream>>>(pooled, G * 128);
  k_zero<<<1, TPB, 0, stream>>>(gcnt, G);
  k_pool_scatter<<<CDIV(N * 32, TPB), TPB, 0, stream>>>(hgB, hsB, batch, pooled, gcnt, N);
  k_pool_norm<<<CDIV(G * 128, TPB), TPB, 0, stream>>>(pooled, gcnt, G);

  // ---- 7. MLP head
  k_head<<<G, 64, 0, stream>>>(pooled, fc1_W, fc1_b, bn1, fc2_W, fc2_b, bn2,
                               fc3_W, fc3_b, (float*)d_out);
}
